// DepthShareGPT_5652176961682
// MI455X (gfx1250) — matrix operand order repaired and checked by reading, not compiled, not run on hardware
//
#include <hip/hip_runtime.h>
#include <stdint.h>

// ---------------- model constants ----------------
#define T_SEQ   1024
#define EMB     1024
#define NH      16
#define NKV     4
#define DH      64
#define KVDIM   256      // NKV*DH
#define NLAYER  2
#define NREP    2
#define VOCAB   32000
#define FF      4096

typedef __bf16 bf16_t;
typedef bf16_t v16bf __attribute__((ext_vector_type(16)));
typedef float  v8f   __attribute__((ext_vector_type(8)));

struct __align__(16) U128 { uint32_t x[4]; };
union FragU { U128 h[2]; v16bf v; };

__device__ inline unsigned short f2bf(float f) {
  uint32_t u = __float_as_uint(f);
  u += 0x7FFFu + ((u >> 16) & 1u);          // round-to-nearest-even
  return (unsigned short)(u >> 16);
}

// CDNA5 async copy: global -> LDS directly, tracked by ASYNCcnt (no VGPR
// round-trip, no loadcnt/dscnt serialization in the staging path).
__device__ __forceinline__ void async_load_b128(unsigned lds_off, const unsigned short* gptr) {
  asm volatile("global_load_async_to_lds_b128 %0, %1, off"
               :
               : "v"(lds_off), "v"(gptr)
               : "memory");
}
__device__ __forceinline__ void wait_async0() {
  asm volatile("s_wait_asynccnt 0x0" ::: "memory");
}

// ---------------- WMMA GEMM: C[M,N] (f32) = A[M,K](bf16,row-major) x B, with
// B given transposed as Bt[N][K] (bf16). grid.z batches with per-operand
// element strides; head-group divisors support GQA (h/4 -> kv head).
// Double-buffered LDS tiles filled via global_load_async_to_lds_b128. -------
__global__ __launch_bounds__(256)
void gemm_bf16_wmma(const unsigned short* __restrict__ A, int lda, long long strideAz, int aDiv,
                    const unsigned short* __restrict__ Bt, int ldb, long long strideBz, int bDiv,
                    float* __restrict__ C, int ldc, long long strideCz,
                    int N, int K, int accumulate)
{
  const int z = blockIdx.z;
  A  += (long long)(z / aDiv) * strideAz;
  Bt += (long long)(z / bDiv) * strideBz;
  C  += (long long)z * strideCz;

  const int tid  = threadIdx.x;
  const int lane = tid & 31;
  const int wave = tid >> 5;
  const int wm   = (wave & 1) * 64;   // wave M offset inside 128
  const int wn   = (wave >> 1) * 32;  // wave N offset inside 128
  const int m0   = blockIdx.y * 128;
  const int n0   = blockIdx.x * 128;

  __shared__ __align__(16) unsigned short As[2][128][40];
  __shared__ __align__(16) unsigned short Bs[2][128][40];

  // Zero-fill B rows beyond N once (only hit for the N=64 att@V GEMM). These
  // rows are never touched by the async loads, so they stay zero for all k.
  if (n0 + 128 > N) {
#pragma unroll
    for (int i = 0; i < 2; ++i) {
      int c = tid + 256 * i, row = c >> 2, kc = (c & 3) * 8;
      if (n0 + row >= N) {
        U128 zchunk = {{0u,0u,0u,0u}};
        *(U128*)&Bs[0][row][kc] = zchunk;
        *(U128*)&Bs[1][row][kc] = zchunk;
      }
    }
  }

  // issue one 128x32 bf16 tile of A and of Bt into LDS buffer `buf`
  auto issue_tile = [&](int k0, int buf) {
#pragma unroll
    for (int i = 0; i < 2; ++i) {
      int c = tid + 256 * i, row = c >> 2, kc = (c & 3) * 8;
      async_load_b128((unsigned)(size_t)&As[buf][row][kc],
                      A + (size_t)(m0 + row) * lda + (k0 + kc));
      int nrow = n0 + row;
      if (nrow < N)
        async_load_b128((unsigned)(size_t)&Bs[buf][row][kc],
                        Bt + (size_t)nrow * ldb + (k0 + kc));
    }
  };

  v8f acc[4][2];
  const v8f vzero = {0.f,0.f,0.f,0.f,0.f,0.f,0.f,0.f};
#pragma unroll
  for (int i = 0; i < 4; ++i)
#pragma unroll
    for (int j = 0; j < 2; ++j) acc[i][j] = vzero;

  issue_tile(0, 0);
  wait_async0();
  __syncthreads();

  int cur = 0;
  for (int k0 = 0; k0 < K; k0 += 32) {
    if (k0 + 32 < K) issue_tile(k0 + 32, cur ^ 1);   // DMA next tile, overlapped

    // fragment loads follow CDNA5 16-bit A/B VGPR layouts (ISA 7.12.2)
    FragU fa[4], fb[2];
    const bool lo = (lane < 16);
#pragma unroll
    for (int mt = 0; mt < 4; ++mt) {
      int mr = wm + mt * 16 + (lane & 15);
      int ka = lo ? 0 : 8;
      fa[mt].h[0] = *(const U128*)&As[cur][mr][ka];       // K = {0..7 | 8..15}
      fa[mt].h[1] = *(const U128*)&As[cur][mr][ka + 16];  // K = {16..23 | 24..31}
    }
#pragma unroll
    for (int nt = 0; nt < 2; ++nt) {
      int nr = wn + nt * 16 + (lane & 15);
      int kb = lo ? 0 : 8;
      fb[nt].h[0] = *(const U128*)&Bs[cur][nr][kb];       // K = {0..7 | 8..15}
      fb[nt].h[1] = *(const U128*)&Bs[cur][nr][kb + 16];  // K = {16..23 | 24..31}
    }
#pragma unroll
    for (int mt = 0; mt < 4; ++mt)
#pragma unroll
      for (int nt = 0; nt < 2; ++nt)
        acc[mt][nt] = __builtin_amdgcn_wmma_f32_16x16x32_bf16(
            false, fa[mt].v, false, fb[nt].v, (short)0, acc[mt][nt], false, false);

    wait_async0();        // next tile fully landed in LDS
    __syncthreads();      // all waves done reading `cur`, next buffer visible
    cur ^= 1;
  }

  // C/D layout: element r -> row = base + r + (lane<16?0:8), col = lane&15
#pragma unroll
  for (int mt = 0; mt < 4; ++mt) {
#pragma unroll
    for (int nt = 0; nt < 2; ++nt) {
      int col = n0 + wn + nt * 16 + (lane & 15);
      if (col >= N) continue;
      int rowBase = m0 + wm + mt * 16 + ((lane < 16) ? 0 : 8);
#pragma unroll
      for (int r = 0; r < 8; ++r) {
        float* cp = C + (size_t)(rowBase + r) * ldc + col;
        float v = acc[mt][nt][r];
        if (accumulate) v += *cp;
        *cp = v;
      }
    }
  }
}

// ---------------- LayerNorm (f32 in, bf16 out), one row per block ----------
__global__ __launch_bounds__(256)
void layernorm_bf16(const float* __restrict__ x, const float* __restrict__ g,
                    const float* __restrict__ b, unsigned short* __restrict__ out)
{
  int t = blockIdx.x;
  const float* row = x + (size_t)t * EMB;
  __shared__ float s1[256], s2[256];
  float v[4], sum = 0.f, sq = 0.f;
#pragma unroll
  for (int i = 0; i < 4; ++i) {
    v[i] = row[threadIdx.x + 256 * i];
    sum += v[i]; sq += v[i] * v[i];
  }
  s1[threadIdx.x] = sum; s2[threadIdx.x] = sq; __syncthreads();
  for (int s = 128; s > 0; s >>= 1) {
    if (threadIdx.x < s) { s1[threadIdx.x] += s1[threadIdx.x + s]; s2[threadIdx.x] += s2[threadIdx.x + s]; }
    __syncthreads();
  }
  float mu = s1[0] * (1.f / EMB);
  float var = s2[0] * (1.f / EMB) - mu * mu;
  float rs = rsqrtf(var + 1e-5f);
#pragma unroll
  for (int i = 0; i < 4; ++i) {
    int e = threadIdx.x + 256 * i;
    out[(size_t)t * EMB + e] = f2bf((v[i] - mu) * rs * g[e] + b[e]);
  }
}

// ---------------- scale + causal mask + softmax + sigmoid-gate threshold ---
__global__ __launch_bounds__(256)
void softmax_gate_bf16(const float* __restrict__ S, const float* __restrict__ gate,
                       unsigned short* __restrict__ attb)
{
  int t = blockIdx.x, h = blockIdx.y;
  const float* row = S + ((size_t)h * T_SEQ + t) * T_SEQ;
  unsigned short* orow = attb + ((size_t)h * T_SEQ + t) * T_SEQ;
  __shared__ float sred[256];
  float v[4], m = -3.4e38f;
#pragma unroll
  for (int i = 0; i < 4; ++i) {
    int j = threadIdx.x + 256 * i;
    v[i] = (j <= t) ? row[j] * 0.125f : -3.4e38f;   // DH^-0.5 = 1/8
    m = fmaxf(m, v[i]);
  }
  sred[threadIdx.x] = m; __syncthreads();
  for (int s = 128; s > 0; s >>= 1) { if (threadIdx.x < s) sred[threadIdx.x] = fmaxf(sred[threadIdx.x], sred[threadIdx.x + s]); __syncthreads(); }
  m = sred[0]; __syncthreads();
  float sum = 0.f;
#pragma unroll
  for (int i = 0; i < 4; ++i) {
    int j = threadIdx.x + 256 * i;
    v[i] = (j <= t) ? __expf(v[i] - m) : 0.f;
    sum += v[i];
  }
  sred[threadIdx.x] = sum; __syncthreads();
  for (int s = 128; s > 0; s >>= 1) { if (threadIdx.x < s) sred[threadIdx.x] += sred[threadIdx.x + s]; __syncthreads(); }
  float inv = 1.f / sred[0];
  float thr = 1.f / (1.f + __expf(-gate[h]));
#pragma unroll
  for (int i = 0; i < 4; ++i) {
    int j = threadIdx.x + 256 * i;
    float p = v[i] * inv;
    if (p < thr) p = 0.f;
    orow[j] = f2bf(p);
  }
}

// ---------------- RoPE on first 32 dims of each 64-dim head, f32->bf16 -----
__global__ void rope_bf16(const float* __restrict__ x, unsigned short* __restrict__ xb, int ncols)
{
  int idx = blockIdx.x * 256 + threadIdx.x;
  int t = idx / ncols;
  int e = idx - t * ncols;
  int d = e & 63;
  float val = x[idx], out;
  if (d < 32) {
    int dd = d & 15;
    float theta = __powf(10000.f, -(float)dd * (1.f / 16.f));
    float a = (float)t * theta;
    float sn, cs; __sincosf(a, &sn, &cs);
    float partner = (d < 16) ? -x[idx + 16] : x[idx - 16];  // rotate_half
    out = val * cs + partner * sn;
  } else out = val;
  xb[idx] = f2bf(out);
}

// ---------------- V transpose per kv head: vtb[kvh][d][s] = v[s][kvh*64+d] -
__global__ void vtrans_bf16(const float* __restrict__ v, unsigned short* __restrict__ vtb)
{
  int idx = blockIdx.x * 256 + threadIdx.x;     // NKV*DH*T
  int s = idx & (T_SEQ - 1);
  int rem = idx >> 10;
  int d = rem & 63;
  int kvh = rem >> 6;
  vtb[idx] = f2bf(v[(size_t)s * KVDIM + kvh * DH + d]);
}

__global__ void gather_embed(const float* __restrict__ embed, const int* __restrict__ ids,
                             float* __restrict__ x)
{
  int idx = blockIdx.x * 256 + threadIdx.x;
  int t = idx >> 10;
  int e = idx & 1023;
  x[idx] = embed[(size_t)ids[t] * EMB + e];
}

__global__ void conv_bf16(const float* __restrict__ s, unsigned short* __restrict__ d, int n)
{ int i = blockIdx.x * 256 + threadIdx.x; if (i < n) d[i] = f2bf(s[i]); }

// dst[n*K + k] = src[k*N + n]   (f32 [K,N] -> bf16 [N,K])
__global__ void convT_bf16(const float* __restrict__ s, unsigned short* __restrict__ d, int K, int N)
{
  long long i = (long long)blockIdx.x * 256 + threadIdx.x;
  if (i < (long long)K * N) {
    int n = (int)(i / K);
    int k = (int)(i - (long long)n * K);
    d[i] = f2bf(s[(size_t)k * N + n]);
  }
}

__global__ void gelu_bf16(const float* __restrict__ s, unsigned short* __restrict__ d, int n)
{
  int i = blockIdx.x * 256 + threadIdx.x;
  if (i < n) { float x = s[i]; d[i] = f2bf(0.5f * x * (1.f + erff(x * 0.70710678118f))); }
}

// ---------------- cross-entropy: per-row logsumexp then mean ---------------
__global__ __launch_bounds__(256)
void loss_row(const float* __restrict__ logits, const int* __restrict__ tgt,
              float* __restrict__ row_loss)
{
  int t = blockIdx.x;
  const float* row = logits + (size_t)t * VOCAB;
  __shared__ float sred[256];
  float m = -3.4e38f;
  for (int j = threadIdx.x; j < VOCAB; j += 256) m = fmaxf(m, row[j]);
  sred[threadIdx.x] = m; __syncthreads();
  for (int s = 128; s > 0; s >>= 1) { if (threadIdx.x < s) sred[threadIdx.x] = fmaxf(sred[threadIdx.x], sred[threadIdx.x + s]); __syncthreads(); }
  m = sred[0]; __syncthreads();
  float sum = 0.f;
  for (int j = threadIdx.x; j < VOCAB; j += 256) sum += __expf(row[j] - m);
  sred[threadIdx.x] = sum; __syncthreads();
  for (int s = 128; s > 0; s >>= 1) { if (threadIdx.x < s) sred[threadIdx.x] += sred[threadIdx.x + s]; __syncthreads(); }
  if (threadIdx.x == 0) {
    float lse = m + __logf(sred[0]);
    row_loss[t] = lse - row[tgt[t]];
  }
}

__global__ __launch_bounds__(256)
void loss_reduce(const float* __restrict__ row_loss, float* __restrict__ out)
{
  __shared__ float sred[256];
  float s = 0.f;
  for (int j = threadIdx.x; j < T_SEQ; j += 256) s += row_loss[j];
  sred[threadIdx.x] = s; __syncthreads();
  for (int k = 128; k > 0; k >>= 1) { if (threadIdx.x < k) sred[threadIdx.x] += sred[threadIdx.x + k]; __syncthreads(); }
  if (threadIdx.x == 0) out[0] = sred[0] * (1.f / (float)T_SEQ);
}

// ---------------- orchestration ----------------
extern "C" void kernel_launch(void* const* d_in, const int* in_sizes, int n_in,
                              void* d_out, int out_size, void* d_ws, size_t ws_size,
                              hipStream_t stream)
{
  (void)in_sizes; (void)n_in; (void)out_size; (void)ws_size;
  const float* embed = (const float*)d_in[0];
  const float* ln1_g = (const float*)d_in[1];
  const float* ln1_b = (const float*)d_in[2];
  const float* Wq    = (const float*)d_in[3];
  const float* Wk    = (const float*)d_in[4];
  const float* Wv    = (const float*)d_in[5];
  const float* Wo    = (const float*)d_in[6];
  const float* gate  = (const float*)d_in[7];
  const float* ln2_g = (const float*)d_in[8];
  const float* ln2_b = (const float*)d_in[9];
  const float* W1    = (const float*)d_in[10];
  const float* W2    = (const float*)d_in[11];
  const float* lnf_g = (const float*)d_in[12];
  const float* lnf_b = (const float*)d_in[13];
  const int*   ids   = (const int*)d_in[14];
  const int*   tgt   = (const int*)d_in[15];
  float* logits = (float*)d_out;                       // [T, V]
  float* lossp  = logits + (size_t)T_SEQ * VOCAB;      // scalar

  char* wp = (char*)d_ws;
  auto alloc = [&](size_t bytes) -> void* {
    void* p = (void*)wp; wp += (bytes + 255) & ~(size_t)255; return p;
  };

  unsigned short* embB = (unsigned short*)alloc((size_t)VOCAB * EMB * 2);
  unsigned short *WqT[NLAYER], *WkT[NLAYER], *WvT[NLAYER], *WoT[NLAYER], *W1T[NLAYER], *W2T[NLAYER];
  for (int l = 0; l < NLAYER; ++l) {
    WqT[l] = (unsigned short*)alloc((size_t)EMB * EMB * 2);
    WkT[l] = (unsigned short*)alloc((size_t)KVDIM * EMB * 2);
    WvT[l] = (unsigned short*)alloc((size_t)KVDIM * EMB * 2);
    WoT[l] = (unsigned short*)alloc((size_t)EMB * EMB * 2);
    W1T[l] = (unsigned short*)alloc((size_t)FF * EMB * 2);
    W2T[l] = (unsigned short*)alloc((size_t)EMB * FF * 2);
  }
  float* x            = (float*)alloc((size_t)T_SEQ * EMB * 4);
  unsigned short* hb  = (unsigned short*)alloc((size_t)T_SEQ * EMB * 2);
  float* q            = (float*)alloc((size_t)T_SEQ * EMB * 4);
  float* kf           = (float*)alloc((size_t)T_SEQ * KVDIM * 4);
  float* vf           = (float*)alloc((size_t)T_SEQ * KVDIM * 4);
  unsigned short* qb  = (unsigned short*)alloc((size_t)T_SEQ * EMB * 2);
  unsigned short* kb  = (unsigned short*)alloc((size_t)T_SEQ * KVDIM * 2);
  unsigned short* vtb = (unsigned short*)alloc((size_t)NKV * DH * T_SEQ * 2);
  float* S            = (float*)alloc((size_t)NH * T_SEQ * T_SEQ * 4);
  unsigned short* attb= (unsigned short*)alloc((size_t)NH * T_SEQ * T_SEQ * 2);
  float* y            = (float*)alloc((size_t)T_SEQ * EMB * 4);
  unsigned short* yb  = (unsigned short*)alloc((size_t)T_SEQ * EMB * 2);
  float* a1           = (float*)alloc((size_t)T_SEQ * FF * 4);
  unsigned short* gb  = (unsigned short*)alloc((size_t)T_SEQ * FF * 2);
  float* row_loss     = (float*)alloc((size_t)T_SEQ * 4);

  // ---- weight prep (per call; no caching allowed) ----
  conv_bf16<<<(VOCAB * EMB + 255) / 256, 256, 0, stream>>>(embed, embB, VOCAB * EMB);
  for (int l = 0; l < NLAYER; ++l) {
    convT_bf16<<<(EMB * EMB + 255) / 256, 256, 0, stream>>>(Wq + (size_t)l * EMB * EMB,   WqT[l], EMB, EMB);
    convT_bf16<<<(EMB * KVDIM + 255) / 256, 256, 0, stream>>>(Wk + (size_t)l * EMB * KVDIM, WkT[l], EMB, KVDIM);
    convT_bf16<<<(EMB * KVDIM + 255) / 256, 256, 0, stream>>>(Wv + (size_t)l * EMB * KVDIM, WvT[l], EMB, KVDIM);
    convT_bf16<<<(EMB * EMB + 255) / 256, 256, 0, stream>>>(Wo + (size_t)l * EMB * EMB,   WoT[l], EMB, EMB);
    convT_bf16<<<(EMB * FF + 255) / 256, 256, 0, stream>>>(W1 + (size_t)l * EMB * FF,     W1T[l], EMB, FF);
    convT_bf16<<<(EMB * FF + 255) / 256, 256, 0, stream>>>(W2 + (size_t)l * FF * EMB,     W2T[l], FF, EMB);
  }

  gather_embed<<<(T_SEQ * EMB) / 256, 256, 0, stream>>>(embed, ids, x);

  for (int r = 0; r < NREP; ++r)
  for (int l = 0; l < NLAYER; ++l) {
    layernorm_bf16<<<T_SEQ, 256, 0, stream>>>(x, ln1_g + l * EMB, ln1_b + l * EMB, hb);
    gemm_bf16_wmma<<<dim3(EMB / 128, T_SEQ / 128, 1), 256, 0, stream>>>(
        hb, EMB, 0, 1, WqT[l], EMB, 0, 1, q, EMB, 0, EMB, EMB, 0);
    gemm_bf16_wmma<<<dim3(KVDIM / 128, T_SEQ / 128, 1), 256, 0, stream>>>(
        hb, EMB, 0, 1, WkT[l], EMB, 0, 1, kf, KVDIM, 0, KVDIM, EMB, 0);
    gemm_bf16_wmma<<<dim3(KVDIM / 128, T_SEQ / 128, 1), 256, 0, stream>>>(
        hb, EMB, 0, 1, WvT[l], EMB, 0, 1, vf, KVDIM, 0, KVDIM, EMB, 0);
    rope_bf16<<<(T_SEQ * EMB) / 256, 256, 0, stream>>>(q, qb, EMB);
    rope_bf16<<<(T_SEQ * KVDIM) / 256, 256, 0, stream>>>(kf, kb, KVDIM);
    vtrans_bf16<<<(NKV * DH * T_SEQ) / 256, 256, 0, stream>>>(vf, vtb);
    // S[h] = Q_h (A, col-offset h*64) @ K_h^T (kb rows are natural [s][d] = Bt)
    gemm_bf16_wmma<<<dim3(T_SEQ / 128, T_SEQ / 128, NH), 256, 0, stream>>>(
        qb, EMB, DH, 1, kb, KVDIM, DH, NH / NKV,
        S, T_SEQ, (long long)T_SEQ * T_SEQ, T_SEQ, DH, 0);
    softmax_gate_bf16<<<dim3(T_SEQ, NH), 256, 0, stream>>>(S, gate + l * NH, attb);
    // Y_h = att_h @ V_h  (Bt = vtb[kvh] is V^T [d][s])
    gemm_bf16_wmma<<<dim3(1, T_SEQ / 128, NH), 256, 0, stream>>>(
        attb, T_SEQ, (long long)T_SEQ * T_SEQ, 1, vtb, T_SEQ, (long long)DH * T_SEQ, NH / NKV,
        y, EMB, DH, DH, T_SEQ, 0);
    conv_bf16<<<(T_SEQ * EMB) / 256, 256, 0, stream>>>(y, yb, T_SEQ * EMB);
    gemm_bf16_wmma<<<dim3(EMB / 128, T_SEQ / 128, 1), 256, 0, stream>>>(   // x += y @ Wo
        yb, EMB, 0, 1, WoT[l], EMB, 0, 1, x, EMB, 0, EMB, EMB, 1);
    layernorm_bf16<<<T_SEQ, 256, 0, stream>>>(x, ln2_g + l * EMB, ln2_b + l * EMB, hb);
    gemm_bf16_wmma<<<dim3(FF / 128, T_SEQ / 128, 1), 256, 0, stream>>>(
        hb, EMB, 0, 1, W1T[l], EMB, 0, 1, a1, FF, 0, FF, EMB, 0);
    gelu_bf16<<<(T_SEQ * FF) / 256, 256, 0, stream>>>(a1, gb, T_SEQ * FF);
    gemm_bf16_wmma<<<dim3(EMB / 128, T_SEQ / 128, 1), 256, 0, stream>>>(   // x += gelu @ W2
        gb, FF, 0, 1, W2T[l], FF, 0, 1, x, EMB, 0, EMB, FF, 1);
  }

  layernorm_bf16<<<T_SEQ, 256, 0, stream>>>(x, lnf_g, lnf_b, hb);
  // logits = x @ embed^T : embB row-major [V][E] is exactly Bt [N][K]
  gemm_bf16_wmma<<<dim3(VOCAB / 128, T_SEQ / 128, 1), 256, 0, stream>>>(
      hb, EMB, 0, 1, embB, EMB, 0, 1, logits, VOCAB, 0, VOCAB, EMB, 0);
  loss_row<<<T_SEQ, 256, 0, stream>>>(logits, tgt, row_loss);
  loss_reduce<<<1, 256, 0, stream>>>(row_loss, lossp);
}
